// pointwise_attention_80126909874717
// MI455X (gfx1250) — matrix operand order repaired and checked by reading, not compiled, not run on hardware
//
#include <hip/hip_runtime.h>
#include <hip/hip_bf16.h>

// ---------------------------------------------------------------------------
// Bahdanau additive attention for MI455X (gfx1250, wave32, WMMA).
// B=32, T=2048, D=512, U=512.
// Strategy: pre-pack h_enc and W1_enc into fragment-linear bf16 so the GEMM
// kernel issues pure global_load_b128 -> v_wmma with zero staging VALU.
// ---------------------------------------------------------------------------

#define B_  32
#define T_  2048
#define D_  512
#define U_  512

typedef __attribute__((ext_vector_type(16))) __bf16 v16bf;
typedef __attribute__((ext_vector_type(8)))  float  v8f;

// f32 -> bf16 round-to-nearest-even via bit ops.
static __device__ inline __bf16 f2bf(float f) {
    union { float f; unsigned u; } v;
    v.f = f;
    unsigned r = v.u + 0x7FFFu + ((v.u >> 16) & 1u);
    unsigned short h = (unsigned short)(r >> 16);
    __bf16 out;
    __builtin_memcpy(&out, &h, 2);
    return out;
}

// ---------------------------------------------------------------------------
// Pack W1_enc (512x512 f32, row-major [k][n]) into B-fragment-linear bf16:
//   Bpack[(s*512 + n)*32 + half*16 + i] = bf16(W1[(s*32 + 8*half + i%8 + 16*(i/8))*512 + n])
// A lane with column n and K-half `half` reads 16 contiguous bf16 (32 B).
// ---------------------------------------------------------------------------
__global__ void k_pack_w1(const float* __restrict__ W1,
                          __bf16* __restrict__ Bpack) {
    int idx = blockIdx.x * blockDim.x + threadIdx.x;    // [0, 512*512)
    int n = idx & 511;
    int k = idx >> 9;
    int s    = k >> 5;
    int k32  = k & 31;
    int half = (k32 >> 3) & 1;
    int i    = (k32 & 7) + ((k32 >> 4) << 3);
    Bpack[((size_t)(s * U_ + n)) * 32 + half * 16 + i] = f2bf(W1[(size_t)k * U_ + n]);
}

// ---------------------------------------------------------------------------
// Pack h_enc (B,T,D f32) into A-fragment-linear bf16.
// A tile = 16 rows (tt) x 32 K (s). Wave32 A layout for 16x16x32 bf16:
//   lane = m + (hi8 ? 16 : 0), elem = (k32 & 7) + (half ? 8 : 0)
//   where k32 = k % 32, hi8 = (k32>>3)&1, half = k32>>4, m = t % 16.
//   Apack[(((b*128 + tt)*16 + s)*32 + lane)*16 + elem]
// ---------------------------------------------------------------------------
__global__ void k_pack_a(const float* __restrict__ h_enc,
                         __bf16* __restrict__ Apack) {
    size_t idx = (size_t)blockIdx.x * blockDim.x + threadIdx.x; // [0, B*T*D)
    int k = (int)(idx & (D_ - 1));
    int t = (int)((idx >> 9) & (T_ - 1));
    int b = (int)(idx >> 20);               // 512*2048 = 2^20
    int m    = t & 15;
    int tt   = t >> 4;
    int s    = k >> 5;
    int k32  = k & 31;
    int hi8  = (k32 >> 3) & 1;
    int half = k32 >> 4;
    int lane = m + (hi8 << 4);
    int elem = (k32 & 7) + (half << 3);
    size_t dst = ((((size_t)b * (T_ / 16) + tt) * 16 + s) * 32 + lane) * 16 + elem;
    Apack[dst] = f2bf(h_enc[idx]);
}

// ---------------------------------------------------------------------------
// dec_proj[b][u] = b1[u] + sum_d h_dec_prev[b][d] * W1[D_ + d][u]
// ---------------------------------------------------------------------------
__global__ void k_dec_proj(const float* __restrict__ hdec,
                           const float* __restrict__ W1,
                           const float* __restrict__ b1,
                           float* __restrict__ dec_proj) {
    int idx = blockIdx.x * blockDim.x + threadIdx.x;   // [0, B_*U_)
    int b = idx / U_;
    int u = idx % U_;
    float acc = b1[u];
    const float* hrow = hdec + (size_t)b * D_;
    const float* wcol = W1 + (size_t)D_ * U_ + u;
    #pragma unroll 8
    for (int d = 0; d < D_; ++d) {
        acc = fmaf(hrow[d], wcol[(size_t)d * U_], acc);
    }
    dec_proj[idx] = acc;
}

// ---------------------------------------------------------------------------
// Fused scores kernel. Workgroup = 128 T-rows x 256 of the 512 U cols, 8 waves laid out
// 4(M) x 2(N). Each wave: 2 M-tiles x 8 N-tiles = 16 WMMA accumulators,
// fragments loaded straight from the packed global buffers (32 B per lane).
// ---------------------------------------------------------------------------
__global__ __launch_bounds__(256) void k_scores(
        const __bf16* __restrict__ Apack,
        const __bf16* __restrict__ Bpack,
        const float* __restrict__ dec_proj,
        const float* __restrict__ W2,
        const float* __restrict__ b2,
        float* __restrict__ scores) {

    __shared__ float lds_scores[128];

    const int b   = blockIdx.x;
    const int t0  = blockIdx.y * 128;
    const int tid = threadIdx.x;
    const int wave = tid >> 5;
    const int lane = tid & 31;
    const int ln16 = lane & 15;
    const int w_m = wave & 3;        // 0..3 -> 32-row group
    const int w_n = wave >> 2;       // 0..1 -> 256-col group
    const int hiB = (lane >> 4) & 1; // B-frag K-half select

    if (tid < 128) lds_scores[tid] = 0.0f;
    __syncthreads();

    v8f acc[2][8];
    #pragma unroll
    for (int mi = 0; mi < 2; ++mi)
        #pragma unroll
        for (int nt = 0; nt < 8; ++nt) acc[mi][nt] = (v8f){};

    const v16bf* Av = (const v16bf*)Apack;
    const v16bf* Bv = (const v16bf*)Bpack;

    // A fragment base (in v16bf units): (((b*128 + tt)*16 + s)*32 + lane)
    const int tt0 = blockIdx.y * 8 + w_m * 2;   // tile-row for mi=0
    const size_t abase0 = (((size_t)b * (T_ / 16) + tt0) * 16) * 32 + lane;
    const size_t abase1 = abase0 + 16 * 32;     // mi=1 is next 16-row tile
    // B fragment (in v16bf units): (s*512 + n)*2 + hiB, n = w_n*256 + nt*16 + ln16
    const size_t bbase = (size_t)(w_n * 256 + ln16) * 2 + hiB;

    for (int s = 0; s < 16; ++s) {
        v16bf a0 = Av[abase0 + (size_t)s * 32];
        v16bf a1 = Av[abase1 + (size_t)s * 32];
        const size_t brow = bbase + (size_t)s * U_ * 2;
        #pragma unroll
        for (int nt = 0; nt < 8; ++nt) {
            v16bf bf = Bv[brow + (size_t)nt * 32];   // (nt*16 cols) * 2
            acc[0][nt] = __builtin_amdgcn_wmma_f32_16x16x32_bf16(
                false, a0, false, bf, (short)0, acc[0][nt], false, false);
            acc[1][nt] = __builtin_amdgcn_wmma_f32_16x16x32_bf16(
                false, a1, false, bf, (short)0, acc[1][nt], false, false);
        }
    }

    // Epilogue: + dec_proj (b1 folded), tanh, dot with W2, per-row reduce.
    float dp[8], w2v[8];
    #pragma unroll
    for (int nt = 0; nt < 8; ++nt) {
        const int u = w_n * 256 + nt * 16 + ln16;
        dp[nt]  = dec_proj[(size_t)b * U_ + u];
        w2v[nt] = W2[u];
    }
    #pragma unroll
    for (int mi = 0; mi < 2; ++mi) {
        #pragma unroll
        for (int v = 0; v < 8; ++v) {
            float p = 0.0f;
            #pragma unroll
            for (int nt = 0; nt < 8; ++nt) {
                p = fmaf(tanhf(acc[mi][nt][v] + dp[nt]), w2v[nt], p);
            }
            #pragma unroll
            for (int off = 1; off < 16; off <<= 1) p += __shfl_xor(p, off, 32);
            const int row = w_m * 32 + mi * 16 + v + ((lane < 16) ? 0 : 8);
            if (ln16 == 0) atomicAdd(&lds_scores[row], p);
        }
    }
    __syncthreads();

    if (tid < 128) {
        float sc = lds_scores[tid] + b2[0];
        scores[(size_t)b * T_ + t0 + tid] = fmaxf(sc, 0.0f);
    }
}

// ---------------------------------------------------------------------------
// Per-batch softmax over T; writes attn into d_out (+B_*D_ offset).
// ---------------------------------------------------------------------------
__global__ __launch_bounds__(256) void k_softmax(
        const float* __restrict__ scores, float* __restrict__ attn) {
    __shared__ float red[256];
    const int b = blockIdx.x;
    const int tid = threadIdx.x;
    float s[T_ / 256];
    float mx = -1e30f;
    #pragma unroll
    for (int i = 0; i < T_ / 256; ++i) {
        s[i] = scores[(size_t)b * T_ + i * 256 + tid];
        mx = fmaxf(mx, s[i]);
    }
    red[tid] = mx;
    __syncthreads();
    for (int off = 128; off > 0; off >>= 1) {
        if (tid < off) red[tid] = fmaxf(red[tid], red[tid + off]);
        __syncthreads();
    }
    mx = red[0];
    __syncthreads();

    float e[T_ / 256], sum = 0.0f;
    #pragma unroll
    for (int i = 0; i < T_ / 256; ++i) {
        e[i] = __expf(s[i] - mx);
        sum += e[i];
    }
    red[tid] = sum;
    __syncthreads();
    for (int off = 128; off > 0; off >>= 1) {
        if (tid < off) red[tid] += red[tid + off];
        __syncthreads();
    }
    const float inv = 1.0f / red[0];
    #pragma unroll
    for (int i = 0; i < T_ / 256; ++i) {
        attn[(size_t)b * T_ + i * 256 + tid] = e[i] * inv;
    }
}

// ---------------------------------------------------------------------------
// Context partials over 16 T-chunks (deterministic, no atomics).
// ---------------------------------------------------------------------------
__global__ __launch_bounds__(512) void k_ctx_partial(
        const float* __restrict__ h_enc, const float* __restrict__ attn,
        float* __restrict__ partials) {
    const int b = blockIdx.x;
    const int chunk = blockIdx.y;
    const int d = threadIdx.x;           // 0..511
    const int t0 = chunk * (T_ / 16);
    float acc = 0.0f;
    for (int t = t0; t < t0 + (T_ / 16); ++t) {
        acc = fmaf(attn[(size_t)b * T_ + t],
                   h_enc[((size_t)b * T_ + t) * D_ + d], acc);
    }
    partials[((size_t)chunk * B_ + b) * D_ + d] = acc;
}

__global__ void k_ctx_reduce(const float* __restrict__ partials,
                             float* __restrict__ context) {
    int idx = blockIdx.x * blockDim.x + threadIdx.x;  // [0, B_*D_)
    float acc = 0.0f;
    #pragma unroll
    for (int c = 0; c < 16; ++c) acc += partials[(size_t)c * (B_ * D_) + idx];
    context[idx] = acc;
}

// ---------------------------------------------------------------------------
extern "C" void kernel_launch(void* const* d_in, const int* in_sizes, int n_in,
                              void* d_out, int out_size, void* d_ws, size_t ws_size,
                              hipStream_t stream) {
    (void)in_sizes; (void)n_in; (void)out_size; (void)ws_size;

    const float* h_enc = (const float*)d_in[0];   // (B,T,D)
    const float* hdec  = (const float*)d_in[1];   // (B,D)
    const float* W1    = (const float*)d_in[2];   // (2D,U)
    const float* b1    = (const float*)d_in[3];   // (U,)
    const float* W2    = (const float*)d_in[4];   // (U,1)
    const float* b2    = (const float*)d_in[5];   // (1,)

    float* context = (float*)d_out;               // B_*D_ floats
    float* attn    = (float*)d_out + B_ * D_;     // B_*T_ floats

    float* ws       = (float*)d_ws;
    float*  scores   = ws;                                   // 65536 f
    float*  dec_proj = ws + B_ * T_;                         // 16384 f
    float*  partials = ws + B_ * T_ + B_ * U_;               // 262144 f
    __bf16* Apack    = (__bf16*)(ws + B_ * T_ + B_ * U_ + 16 * B_ * D_);
    __bf16* Bpack    = Apack + (size_t)B_ * T_ * D_;         // after 64MB Apack

    // 0) one-time operand packing (f32 -> fragment-linear bf16)
    k_pack_w1<<<(D_ * U_) / 256, 256, 0, stream>>>(W1, Bpack);
    k_pack_a<<<(int)(((size_t)B_ * T_ * D_) / 256), 256, 0, stream>>>(h_enc, Apack);

    // 1) decoder projection (+b1 folded)
    k_dec_proj<<<(B_ * U_) / 256, 256, 0, stream>>>(hdec, W1, b1, dec_proj);

    // 2) fused WMMA GEMM -> tanh -> W2 dot -> relu -> scores
    k_scores<<<dim3(B_, T_ / 128), 256, 0, stream>>>(Apack, Bpack, dec_proj,
                                                     W2, b2, scores);

    // 3) softmax over T -> attn (d_out)
    k_softmax<<<B_, 256, 0, stream>>>(scores, attn);

    // 4) context partials, 5) deterministic reduce -> context (d_out)
    k_ctx_partial<<<dim3(B_, 16), 512, 0, stream>>>(h_enc, attn, partials);
    k_ctx_reduce<<<(B_ * D_) / 256, 256, 0, stream>>>(partials, context);
}
